// SimpleGCN_28999619182643
// MI455X (gfx1250) — matrix operand order repaired and checked by reading, not compiled, not run on hardware
//
#include <hip/hip_runtime.h>
#include <hip/hip_bf16.h>
#include <math.h>

typedef __attribute__((ext_vector_type(16))) _Float16 v16h;
typedef __attribute__((ext_vector_type(8)))  _Float16 h8;
typedef __attribute__((ext_vector_type(8)))  float    v8f;

#define BN_EPS 1e-5f

// ---------------------------------------------------------------------------
// WMMA helpers (CDNA5 wave32, V_WMMA_F32_16X16X32_F16)
// ---------------------------------------------------------------------------
__device__ inline v8f wmma_f16f32(v16h a, v16h b, v8f c) {
  // 8 args: (neg_a, A, neg_b, B, c_mod, C, reuse_a, reuse_b)
  return __builtin_amdgcn_wmma_f32_16x16x32_f16(false, a, false, b, (short)0, c,
                                                false, false);
}

// A tile: 16 x K row-major f16 in LDS, K-slab [k0,k0+32). Per the ISA 16-bit
// A-layout, lane (m = L%16, g = L/16) needs halves K = 8g+0..7 and
// K = 16+8g+0..7 -> two contiguous 16B runs -> 2x ds_load_b128.
// Requires lda % 8 == 0 and k0 % 32 == 0 (always true here).
__device__ inline v16h load_A16(const _Float16* A, int lda, int k0) {
  const int lane = threadIdx.x & 31;
  const int m = lane & 15;
  const int g = lane >> 4;
  const _Float16* base = A + m * lda + k0 + 8 * g;
  h8 lo = *(const h8*)(base);
  h8 hi = *(const h8*)(base + 16);
  v16h a;
#pragma unroll
  for (int i = 0; i < 8; ++i) { a[i] = lo[i]; a[8 + i] = hi[i]; }
  return a;
}

// B tile from TRANSPOSED weights Bt[N][K] (f16, global, K padded to mult of
// 32, N padded to mult of 16). Lane (n = L%16, g = L/16) needs K = 8g+0..7
// and K = 16+8g+0..7 of column n -> two 16B runs -> 2x global_load_b128.
__device__ inline v16h load_Bt16(const _Float16* Bt, int ldk, int k0, int n0) {
  const int lane = threadIdx.x & 31;
  const int n = lane & 15;
  const int g = lane >> 4;
  const _Float16* base = Bt + (long)(n0 + n) * ldk + k0 + 8 * g;
  h8 lo = *(const h8*)(base);
  h8 hi = *(const h8*)(base + 16);
  v16h b;
#pragma unroll
  for (int i = 0; i < 8; ++i) { b[i] = lo[i]; b[8 + i] = hi[i]; }
  return b;
}

// ---------------------------------------------------------------------------
// float atomic max/min via integer monotonicity trick
// ---------------------------------------------------------------------------
__device__ inline void atomicMaxF(float* a, float v) {
  if (v >= 0.0f) atomicMax((int*)a, __float_as_int(v));
  else           atomicMin((unsigned int*)a, __float_as_uint(v));
}
__device__ inline void atomicMinF(float* a, float v) {
  if (v >= 0.0f) atomicMin((int*)a, __float_as_int(v));
  else           atomicMax((unsigned int*)a, __float_as_uint(v));
}

// ---------------------------------------------------------------------------
// Utility kernels
// ---------------------------------------------------------------------------
__global__ void k_fill_f32(float* p, float v, long n) {
  long i = (long)blockIdx.x * blockDim.x + threadIdx.x;
  long s = (long)gridDim.x * blockDim.x;
  for (; i < n; i += s) p[i] = v;
}

__global__ void k_cvt_f16(const float* __restrict__ s, _Float16* __restrict__ d,
                          long n) {
  long i = (long)blockIdx.x * blockDim.x + threadIdx.x;
  long st = (long)gridDim.x * blockDim.x;
  for (; i < n; i += st) d[i] = (_Float16)s[i];
}

// Transpose + pad + f16-convert weights:
//   src: [batch][K][Nn] row-major f32;  dst: [batch][Np][Kp] f16,
//   dst[b][n][k] = src[b][k][n], zero outside.
__global__ void k_cvt_tr(const float* __restrict__ src,
                         _Float16* __restrict__ dst, int K, int Nn, int Kp,
                         int Np, int batch) {
  long total = (long)batch * Np * Kp;
  long i = (long)blockIdx.x * blockDim.x + threadIdx.x;
  long st = (long)gridDim.x * blockDim.x;
  for (; i < total; i += st) {
    int b = (int)(i / ((long)Np * Kp));
    int r = (int)(i % ((long)Np * Kp));
    int n = r / Kp;
    int k = r % Kp;
    float v = (n < Nn && k < K) ? src[((long)b * K + k) * Nn + n] : 0.0f;
    dst[i] = (_Float16)v;
  }
}

// Pad biases: src [batch][Nn] -> dst [batch][Np] (f32), zero outside.
__global__ void k_pad_bias(const float* __restrict__ src,
                           float* __restrict__ dst, int Nn, int Np, int batch) {
  int i = blockIdx.x * blockDim.x + threadIdx.x;
  if (i >= batch * Np) return;
  int b = i / Np, n = i % Np;
  dst[i] = (n < Nn) ? src[b * Nn + n] : 0.0f;
}

__global__ void k_degree(const int* __restrict__ dst, float* __restrict__ deg,
                         int E) {
  int i = blockIdx.x * blockDim.x + threadIdx.x;
  if (i < E) atomicAdd(&deg[dst[i]], 1.0f);
}

__global__ void k_avglog(const float* __restrict__ deg, float* __restrict__ acc,
                         int N) {
  int i = blockIdx.x * blockDim.x + threadIdx.x;
  if (i < N) atomicAdd(acc, logf(deg[i] + 1.0f));
}

__global__ void k_scalers(const float* __restrict__ deg,
                          const float* __restrict__ acc, float* __restrict__ amp,
                          float* __restrict__ att, int N) {
  int i = blockIdx.x * blockDim.x + threadIdx.x;
  if (i < N) {
    float avg = acc[0] / (float)N;
    float d = deg[i];
    float ld = logf((d > 1.0f ? d : 1.0f) + 1.0f);
    amp[i] = ld / avg;
    att[i] = avg / ld;
  }
}

__global__ void k_bn_relu(const float* __restrict__ opre,
                          const float* __restrict__ bnS,
                          const float* __restrict__ bnQ,
                          const float* __restrict__ gma,
                          const float* __restrict__ bta, float* __restrict__ hf,
                          _Float16* __restrict__ h16, int N, int fo) {
  long i = (long)blockIdx.x * blockDim.x + threadIdx.x;
  long n = (long)N * fo;
  if (i >= n) return;
  int c = (int)(i % fo);
  float invN = 1.0f / (float)N;
  float mean = bnS[c] * invN;
  float var = bnQ[c] * invN - mean * mean;
  float v = (opre[i] - mean) * rsqrtf(var + BN_EPS) * gma[c] + bta[c];
  v = v > 0.0f ? v : 0.0f;
  hf[i] = v;
  h16[i] = (_Float16)v;
}

__global__ void k_head(const float* __restrict__ h, const float* __restrict__ W,
                       const float* __restrict__ b, float* __restrict__ out,
                       int N, int F) {
  int i = blockIdx.x * blockDim.x + threadIdx.x;
  if (i >= N) return;
  float acc = b[0];
  for (int c = 0; c < F; ++c) acc += h[(long)i * F + c] * W[c];
  out[i] = acc;
}

// ---------------------------------------------------------------------------
// Edge kernel: one 16-edge tile per wave.
//   msg = [h[dst] | h[src] | edge_attr@We+be]  (16 x 3fi, f16 in LDS)
//   per tower t: m1 = relu(msg@preW1[t]+b1); m = m1@preW2[t]+b2  (WMMA)
//   scatter m: atomic sum / sum-sq / max / min into [N, T*fi] buffers at dst
// All matmul dims are multiples of 32/16 here -> fully unguarded operands.
// ---------------------------------------------------------------------------
__global__ void pna_edge_msg(
    const _Float16* __restrict__ h16, const int* __restrict__ srcI,
    const int* __restrict__ dstI, const float* __restrict__ ea,
    const float* __restrict__ We, const float* __restrict__ be,
    const _Float16* __restrict__ W1t, const float* __restrict__ b1,
    const _Float16* __restrict__ W2t, const float* __restrict__ b2,
    float* __restrict__ aggS, float* __restrict__ aggQ,
    float* __restrict__ aggMx, float* __restrict__ aggMn, int E, int fi) {
  extern __shared__ __align__(16) char smem_raw[];
  _Float16* smem = (_Float16*)smem_raw;
  const int wave = threadIdx.x >> 5;
  const int lane = threadIdx.x & 31;
  const int K1 = 3 * fi;
  _Float16* msg = smem + (size_t)wave * (16 * K1 + 16 * fi);
  _Float16* m1 = msg + 16 * K1;
  const int e0 = (blockIdx.x * (blockDim.x >> 5) + wave) * 16;

  // ---- stage msg tile [16 x 3fi]: vectorized row copies + edge encoder ----
  const int hc = fi >> 3;  // 16-byte chunks per feature vector
  for (int r = 0; r < 16; ++r) {
    const int e = e0 + r;
    const bool ok = e < E;
    const int nd = ok ? dstI[e] : 0;
    const int ns = ok ? srcI[e] : 0;
    for (int c = lane; c < 2 * hc; c += 32) {
      h8 v = {};
      if (ok) {
        const int node = (c < hc) ? nd : ns;
        const int cc = (c < hc) ? c : (c - hc);
        v = *(const h8*)(h16 + (long)node * fi + cc * 8);
      }
      *(h8*)(msg + r * K1 + c * 8) = v;
    }
    for (int c = lane; c < fi; c += 32) {
      float acc = 0.0f;
      if (ok) {
        acc = be[c];
#pragma unroll
        for (int j = 0; j < 8; ++j) acc += ea[(long)e * 8 + j] * We[j * fi + c];
      }
      msg[r * K1 + 2 * fi + c] = (_Float16)acc;
    }
  }
  __syncthreads();

  const int ntiles = fi >> 4;
  const int g = lane >> 4, n = lane & 15;
  for (int t = 0; t < 4; ++t) {
    const _Float16* W1tt = W1t + (long)t * fi * K1;  // [fi][3fi] transposed
    const _Float16* W2tt = W2t + (long)t * fi * fi;  // [fi][fi]  transposed
    // m1 = relu(msg @ W1[t] + b1[t])
    for (int nt = 0; nt < ntiles; ++nt) {
      const int n0 = nt * 16;
      v8f c = {};
      for (int k0 = 0; k0 < K1; k0 += 32) {
        c = wmma_f16f32(load_A16(msg, K1, k0), load_Bt16(W1tt, K1, k0, n0), c);
      }
      const float bias = b1[t * fi + n0 + n];
#pragma unroll
      for (int i = 0; i < 8; ++i) {
        float v = c[i] + bias;
        v = v > 0.0f ? v : 0.0f;
        m1[(i + 8 * g) * fi + n0 + n] = (_Float16)v;
      }
    }
    __syncthreads();
    // m = m1 @ W2[t] + b2[t]; scatter to aggregation buffers
    for (int nt = 0; nt < ntiles; ++nt) {
      const int n0 = nt * 16;
      v8f c = {};
      for (int k0 = 0; k0 < fi; k0 += 32) {
        c = wmma_f16f32(load_A16(m1, fi, k0), load_Bt16(W2tt, fi, k0, n0), c);
      }
      const float bias = b2[t * fi + n0 + n];
#pragma unroll
      for (int i = 0; i < 8; ++i) {
        const int e = e0 + i + 8 * g;
        if (e < E) {
          const float v = c[i] + bias;
          const long o = (long)dstI[e] * (4 * fi) + t * fi + n0 + n;
          atomicAdd(&aggS[o], v);
          atomicAdd(&aggQ[o], v * v);
          atomicMaxF(&aggMx[o], v);
          atomicMinF(&aggMn[o], v);
        }
      }
    }
    __syncthreads();
  }
}

// ---------------------------------------------------------------------------
// Node kernel: one 16-node tile per wave.
//   per tower t: Ain = [h | mean,mn,mx,std | *amp | *att]  (16 x 13fi)
//                o1 = relu(Ain@postW1[t]+b1); o2 = o1@postW2[t]+b2 -> Ofull
//   Ofin = Ofull @ linW + lin_b  -> opre + BN sum/sumsq atomics
// postW1/postW2 are transposed + padded (N->ftp=max(ft,16), K->ktp=max(ft,32)),
// biases pre-padded, so all operand loads are unguarded vector loads.
// ---------------------------------------------------------------------------
__global__ void pna_node_post(
    const _Float16* __restrict__ h16, const float* __restrict__ deg,
    const float* __restrict__ amp, const float* __restrict__ att,
    const float* __restrict__ aggS, const float* __restrict__ aggQ,
    const float* __restrict__ aggMx, const float* __restrict__ aggMn,
    const _Float16* __restrict__ PW1t, const float* __restrict__ pb1p,
    const _Float16* __restrict__ PW2t, const float* __restrict__ pb2p,
    const _Float16* __restrict__ LWt, const float* __restrict__ lb,
    float* __restrict__ opre, float* __restrict__ bnS, float* __restrict__ bnQ,
    int N, int fi, int fo, int ft) {
  extern __shared__ __align__(16) char smem_raw[];
  _Float16* smem = (_Float16*)smem_raw;
  const int wave = threadIdx.x >> 5;
  const int lane = threadIdx.x & 31;
  const int KA = 13 * fi;
  const int ftp = ft < 16 ? 16 : ft;   // padded N of post_W1 / post_W2
  const int ktp = ft < 32 ? 32 : ft;   // padded K of post_W2 (= T1 width)
  _Float16* Ain = smem + (size_t)wave * (16 * KA + 16 * ktp + 16 * fo);
  _Float16* T1 = Ain + 16 * KA;
  _Float16* Ofull = T1 + 16 * ktp;
  const int nb = (blockIdx.x * (blockDim.x >> 5) + wave) * 16;
  const int g = lane >> 4, n = lane & 15;
  const int nt1 = ftp >> 4;

  // zero T1 once: padded K columns (>= ft) stay zero across all towers
  for (int i = lane; i < 16 * ktp; i += 32) T1[i] = (_Float16)0.0f;

  for (int t = 0; t < 4; ++t) {
    // ---- stage Ain [16 x 13fi] ----
    const int hc = fi >> 3;
    for (int r = 0; r < 16; ++r) {
      const int node = nb + r;
      const bool ok = node < N;
      for (int c = lane; c < hc; c += 32) {
        h8 v = {};
        if (ok) v = *(const h8*)(h16 + (long)node * fi + c * 8);
        *(h8*)(Ain + r * KA + c * 8) = v;
      }
      const float d = ok ? deg[node] : 0.0f;
      const float cnt = d > 1.0f ? d : 1.0f;
      const float am = ok ? amp[node] : 0.0f;
      const float at = ok ? att[node] : 0.0f;
      _Float16* row = Ain + r * KA + fi;
      for (int f = lane; f < fi; f += 32) {
        float mean = 0.0f, mn = 0.0f, mx = 0.0f, sd = 0.0f;
        if (ok) {
          const long o = (long)node * (4 * fi) + t * fi + f;
          const float s = aggS[o], q = aggQ[o];
          mean = s / cnt;
          float var = q / cnt - mean * mean;
          if (var < 0.0f) var = 0.0f;
          sd = sqrtf(var + BN_EPS);
          mn = d > 0.0f ? aggMn[o] : 0.0f;
          mx = d > 0.0f ? aggMx[o] : 0.0f;
        }
        const float vals[4] = {mean, mn, mx, sd};
#pragma unroll
        for (int kd = 0; kd < 4; ++kd) {
          const float v0 = vals[kd];
          row[kd * fi + f] = (_Float16)v0;
          row[4 * fi + kd * fi + f] = (_Float16)(v0 * am);
          row[8 * fi + kd * fi + f] = (_Float16)(v0 * at);
        }
      }
    }
    __syncthreads();

    const _Float16* W1tt = PW1t + (long)t * ftp * KA;  // [ftp][13fi]
    const _Float16* W2tt = PW2t + (long)t * ftp * ktp; // [ftp][ktp]
    // o1 = relu(Ain @ postW1[t] + pb1[t]) -> T1 (cols >= ft are zero by pad)
    for (int nt = 0; nt < nt1; ++nt) {
      const int n0 = nt * 16;
      v8f c = {};
      for (int k0 = 0; k0 < KA; k0 += 32) {
        c = wmma_f16f32(load_A16(Ain, KA, k0), load_Bt16(W1tt, KA, k0, n0), c);
      }
      const float bias = pb1p[t * ftp + n0 + n];
#pragma unroll
      for (int i = 0; i < 8; ++i) {
        float v = c[i] + bias;
        v = v > 0.0f ? v : 0.0f;
        T1[(i + 8 * g) * ktp + n0 + n] = (_Float16)v;
      }
    }
    __syncthreads();
    // o2 = T1 @ postW2[t] + pb2[t] -> Ofull columns [t*ft, (t+1)*ft)
    for (int nt = 0; nt < nt1; ++nt) {
      const int n0 = nt * 16;
      v8f c = {};
      for (int k0 = 0; k0 < ktp; k0 += 32) {
        c = wmma_f16f32(load_A16(T1, ktp, k0), load_Bt16(W2tt, ktp, k0, n0), c);
      }
      const float bias = pb2p[t * ftp + n0 + n];
      const bool nok = (n0 + n) < ft;  // don't spill into next tower's columns
#pragma unroll
      for (int i = 0; i < 8; ++i) {
        if (nok)
          Ofull[(i + 8 * g) * fo + t * ft + n0 + n] = (_Float16)(c[i] + bias);
      }
    }
    __syncthreads();
  }

  // ---- tower mixing: Ofin = Ofull @ linW + lin_b ----
  for (int nt = 0; nt < (fo >> 4); ++nt) {
    const int n0 = nt * 16;
    v8f c = {};
    for (int k0 = 0; k0 < fo; k0 += 32) {
      c = wmma_f16f32(load_A16(Ofull, fo, k0), load_Bt16(LWt, fo, k0, n0), c);
    }
    const float bias = lb[n0 + n];
#pragma unroll
    for (int i = 0; i < 8; ++i) {
      const int node = nb + i + 8 * g;
      if (node < N) {
        const float v = c[i] + bias;
        opre[(long)node * fo + n0 + n] = v;
        atomicAdd(&bnS[n0 + n], v);
        atomicAdd(&bnQ[n0 + n], v * v);
      }
    }
  }
}

// ---------------------------------------------------------------------------
// Host orchestration
// ---------------------------------------------------------------------------
extern "C" void kernel_launch(void* const* d_in, const int* in_sizes, int n_in,
                              void* d_out, int out_size, void* d_ws,
                              size_t ws_size, hipStream_t stream) {
  (void)n_in; (void)out_size; (void)ws_size;
  const float* x = (const float*)d_in[0];
  const int* ei = (const int*)d_in[1];
  const float* ea = (const float*)d_in[2];
  const float* headW = (const float*)d_in[3 + 4 * 14 + 0];
  const float* headB = (const float*)d_in[3 + 4 * 14 + 1];
  const int N = in_sizes[0] / 32;
  const int E = in_sizes[1] / 2;
  const int* srcI = ei;      // edge_index[0]
  const int* dstI = ei + E;  // edge_index[1]

  static const int FI[4] = {32, 32, 64, 128};
  static const int FO[4] = {32, 64, 128, 256};

  // ---- workspace layout ----
  char* ws = (char*)d_ws;
  size_t off = 0;
  auto take = [&](size_t bytes) -> char* {
    off = (off + 255) & ~(size_t)255;
    char* p = ws + off;
    off += bytes;
    return p;
  };
  float* deg = (float*)take((size_t)N * 4);
  float* amp = (float*)take((size_t)N * 4);
  float* att = (float*)take((size_t)N * 4);
  float* avga = (float*)take(256);
  float* bnS = (float*)take(256 * 4);
  float* bnQ = (float*)take(256 * 4);
  float* pb1p = (float*)take(4 * 64 * 4);  // padded post-MLP biases (f32)
  float* pb2p = (float*)take(4 * 64 * 4);
  float* hbuf = (float*)take((size_t)N * 256 * 4);
  _Float16* h16 = (_Float16*)take((size_t)N * 256 * 2);
  float* opre = (float*)take((size_t)N * 256 * 4);
  float* aggS = (float*)take((size_t)N * 512 * 4);
  float* aggQ = (float*)take((size_t)N * 512 * 4);
  float* aggMx = (float*)take((size_t)N * 512 * 4);
  float* aggMn = (float*)take((size_t)N * 512 * 4);
  _Float16* w16 = (_Float16*)take((size_t)1200000 * 2);

  auto fill = [&](float* p, float v, long n) {
    int blocks = (int)((n + 255) / 256);
    if (blocks > 4096) blocks = 4096;
    k_fill_f32<<<blocks, 256, 0, stream>>>(p, v, n);
  };
  auto cvt = [&](const float* s, _Float16* d, long n) {
    int blocks = (int)((n + 255) / 256);
    if (blocks > 4096) blocks = 4096;
    k_cvt_f16<<<blocks, 256, 0, stream>>>(s, d, n);
  };
  auto cvt_tr = [&](const float* s, _Float16* d, int K, int Nn, int Kp, int Np,
                    int batch) {
    long total = (long)batch * Np * Kp;
    int blocks = (int)((total + 255) / 256);
    if (blocks > 4096) blocks = 4096;
    k_cvt_tr<<<blocks, 256, 0, stream>>>(s, d, K, Nn, Kp, Np, batch);
  };

  // ---- degrees + PNA scalers ----
  fill(deg, 0.0f, N);
  fill(avga, 0.0f, 1);
  k_degree<<<(E + 255) / 256, 256, 0, stream>>>(dstI, deg, E);
  k_avglog<<<(N + 255) / 256, 256, 0, stream>>>(deg, avga, N);
  k_scalers<<<(N + 255) / 256, 256, 0, stream>>>(deg, avga, amp, att, N);

  // ---- h0 (f16) ----
  cvt(x, h16, (long)N * 32);

  for (int l = 0; l < 4; ++l) {
    const int fi = FI[l], fo = FO[l], ft = fo / 4;
    const int ftp = ft < 16 ? 16 : ft;
    const int ktp = ft < 32 ? 32 : ft;
    void* const* pp = d_in + 3 + l * 14;
    const float* p_We = (const float*)pp[0];
    const float* p_be = (const float*)pp[1];
    const float* p_W1 = (const float*)pp[2];
    const float* p_b1 = (const float*)pp[3];
    const float* p_W2 = (const float*)pp[4];
    const float* p_b2 = (const float*)pp[5];
    const float* p_PW1 = (const float*)pp[6];
    const float* p_pb1 = (const float*)pp[7];
    const float* p_PW2 = (const float*)pp[8];
    const float* p_pb2 = (const float*)pp[9];
    const float* p_LW = (const float*)pp[10];
    const float* p_lb = (const float*)pp[11];
    const float* p_g = (const float*)pp[12];
    const float* p_bb = (const float*)pp[13];

    // f16 transposed/padded weight scratch
    const int K1 = 3 * fi, KA = 13 * fi;
    _Float16* w1t = w16;                                    // [4][fi][3fi]
    _Float16* w2t = w1t + (size_t)4 * fi * K1;              // [4][fi][fi]
    _Float16* pw1t = w2t + (size_t)4 * fi * fi;             // [4][ftp][13fi]
    _Float16* pw2t = pw1t + (size_t)4 * ftp * KA;           // [4][ftp][ktp]
    _Float16* lwt = pw2t + (size_t)4 * ftp * ktp;           // [fo][fo]
    cvt_tr(p_W1, w1t, K1, fi, K1, fi, 4);
    cvt_tr(p_W2, w2t, fi, fi, fi, fi, 4);
    cvt_tr(p_PW1, pw1t, KA, ft, KA, ftp, 4);
    cvt_tr(p_PW2, pw2t, ft, ft, ktp, ftp, 4);
    cvt_tr(p_LW, lwt, fo, fo, fo, fo, 1);
    k_pad_bias<<<(4 * ftp + 255) / 256, 256, 0, stream>>>(p_pb1, pb1p, ft, ftp, 4);
    k_pad_bias<<<(4 * ftp + 255) / 256, 256, 0, stream>>>(p_pb2, pb2p, ft, ftp, 4);

    // aggregation + BN accumulators
    fill(aggS, 0.0f, (long)N * 4 * fi);
    fill(aggQ, 0.0f, (long)N * 4 * fi);
    fill(aggMx, -3.0e38f, (long)N * 4 * fi);
    fill(aggMn, 3.0e38f, (long)N * 4 * fi);
    fill(bnS, 0.0f, fo);
    fill(bnQ, 0.0f, fo);

    // edge message + scatter
    int wavesE = (fi >= 128) ? 2 : (fi >= 64 ? 4 : 8);
    int tilesE = (E + 15) / 16;
    int blocksE = (tilesE + wavesE - 1) / wavesE;
    size_t ldsE = (size_t)wavesE * (16 * K1 + 16 * fi) * 2;
    pna_edge_msg<<<blocksE, wavesE * 32, ldsE, stream>>>(
        h16, srcI, dstI, ea, p_We, p_be, w1t, p_b1, w2t, p_b2, aggS, aggQ,
        aggMx, aggMn, E, fi);

    // node post-MLP + tower mixing + BN stats
    int wavesN = (fi >= 128) ? 1 : 2;
    int tilesN = (N + 15) / 16;
    int blocksN = (tilesN + wavesN - 1) / wavesN;
    size_t ldsN = (size_t)wavesN * (16 * KA + 16 * ktp + 16 * fo) * 2;
    pna_node_post<<<blocksN, wavesN * 32, ldsN, stream>>>(
        h16, deg, amp, att, aggS, aggQ, aggMx, aggMn, pw1t, pb1p, pw2t, pb2p,
        lwt, p_lb, opre, bnS, bnQ, N, fi, fo, ft);

    // BN (batch stats) + ReLU -> next h (f32 + f16)
    long nelem = (long)N * fo;
    k_bn_relu<<<(int)((nelem + 255) / 256), 256, 0, stream>>>(
        opre, bnS, bnQ, p_g, p_bb, hbuf, h16, N, fo);
  }

  // ---- final linear head: [N,256] @ [256,1] + b ----
  k_head<<<(N + 255) / 256, 256, 0, stream>>>(hbuf, headW, headB,
                                              (float*)d_out, N, 256);
}
